// MultiheadAttentionWithRoPE_29033978921636
// MI455X (gfx1250) — matrix operand order repaired and checked by reading, not compiled, not run on hardware
//
#include <hip/hip_runtime.h>
#include <hip/hip_bf16.h>
#include <math.h>

// ---------------------------------------------------------------------------
// MI455X / gfx1250: wave32, WMMA 16x16x32 f16->f32, async LDS staging
// ---------------------------------------------------------------------------

typedef __attribute__((ext_vector_type(16))) _Float16 v16h;
typedef __attribute__((ext_vector_type(8)))  _Float16 v8h;
typedef __attribute__((ext_vector_type(4)))  _Float16 v4h;
typedef __attribute__((ext_vector_type(8)))  float    v8f;

#define WMMA(a, b, c) __builtin_amdgcn_wmma_f32_16x16x32_f16( \
    false, (a), false, (b), (short)0, (c), false, false)

static constexpr int Bb = 2;
static constexpr int Tt = 2048;
static constexpr int Cc = 1024;
static constexpr int Hh = 16;
static constexpr int HD = 64;       // head dim
static constexpr int BT = Bb * Tt;  // 4096 rows

// ---------------------------------------------------------------------------
// Fragment loaders. A/B: contiguous k order (same for both, not ISA table order); C/D per ISA layout.
// A (16x32 f16): lane l: row m = l&15; halves h: K = (l>>4)*16 + h
// B (32x16 f16): lane l: column n = l&15; halves h: K = (l>>4)*16 + h
// C/D (16x16 f32): lane l: col n = l&15; VGPR v: row m = v + 8*(l>>4).
// ---------------------------------------------------------------------------
__device__ __forceinline__ v16h load_a_frag(const _Float16* p0, int ld, int lane) {
  int m = lane & 15, hi = lane >> 4;
  const _Float16* p = p0 + (size_t)m * ld + hi * 16;
  union { v16h v; v8h h[2]; } u;
  u.h[0] = *(const v8h*)(p);
  u.h[1] = *(const v8h*)(p + 8);
  return u.v;
}

__device__ __forceinline__ v16h load_b_frag(const _Float16* p0, int ld, int lane) {
  int n = lane & 15, hb = lane >> 4;
  return *(const v16h*)(p0 + (size_t)n * ld + hb * 16);
}

// 16B-per-lane async global->LDS copy (ASYNCcnt-tracked, gfx1250).
__device__ __forceinline__ void async_copy_16B(unsigned int lds_off, const _Float16* g) {
  unsigned long long ga = (unsigned long long)(uintptr_t)g;
  asm volatile("global_load_async_to_lds_b128 %0, %1, off"
               :: "v"(lds_off), "v"(ga) : "memory");
}

// ---------------------------------------------------------------------------
// fp32 -> f16 conversion (vectorized x4)
// ---------------------------------------------------------------------------
__global__ __launch_bounds__(256) void cvt_f32_f16(const float4* __restrict__ src,
                                                   v4h* __restrict__ dst, int n4) {
  int i = blockIdx.x * blockDim.x + threadIdx.x;
  if (i < n4) {
    float4 x = src[i];
    v4h y;
    y.x = (_Float16)x.x; y.y = (_Float16)x.y;
    y.z = (_Float16)x.z; y.w = (_Float16)x.w;
    dst[i] = y;
  }
}

// ---------------------------------------------------------------------------
// C[M,N] = A[M,K] @ W[N,K]^T  (f16 in, f32 out).
// 256 threads = 8 waves; wave (r=wid>>2, c=wid&3) owns 32x32 tile;
// block tile 64 x 128. Both operands are K-contiguous -> direct fragment loads.
// ---------------------------------------------------------------------------
__global__ __launch_bounds__(256) void gemm_xwT(const _Float16* __restrict__ A,
                                                const _Float16* __restrict__ W,
                                                float* __restrict__ Cout,
                                                int M, int N, int K) {
  int lane = threadIdx.x & 31;
  int wid  = threadIdx.x >> 5;
  int m0 = blockIdx.y * 64  + (wid >> 2) * 32;
  int n0 = blockIdx.x * 128 + (wid & 3) * 32;

  v8f acc[2][2] = {};
  for (int k0 = 0; k0 < K; k0 += 32) {
    v16h a0 = load_a_frag(A + (size_t)m0 * K + k0,        K, lane);
    v16h a1 = load_a_frag(A + (size_t)(m0 + 16) * K + k0, K, lane);
    v16h b0 = load_b_frag(W + (size_t)n0 * K + k0,        K, lane);
    v16h b1 = load_b_frag(W + (size_t)(n0 + 16) * K + k0, K, lane);
    acc[0][0] = WMMA(a0, b0, acc[0][0]);
    acc[0][1] = WMMA(a0, b1, acc[0][1]);
    acc[1][0] = WMMA(a1, b0, acc[1][0]);
    acc[1][1] = WMMA(a1, b1, acc[1][1]);
  }

  int n = lane & 15, hi = lane >> 4;
  for (int mi = 0; mi < 2; mi++)
    for (int ni = 0; ni < 2; ni++)
      for (int v = 0; v < 8; v++) {
        int row = m0 + mi * 16 + hi * 8 + v;
        int col = n0 + ni * 16 + n;
        Cout[(size_t)row * N + col] = acc[mi][ni][v];
      }
}

// ---------------------------------------------------------------------------
// RoPE + head reshape.
//   Qr, Kr: (B,H,T,HD) f16 ; Vt: (B,H,HD,T) f16 (transposed for P@V operand)
// ---------------------------------------------------------------------------
__global__ __launch_bounds__(256) void rope_reshape(const float* __restrict__ Q,
                                                    const float* __restrict__ K,
                                                    const float* __restrict__ V,
                                                    const float* __restrict__ freqs,
                                                    _Float16* __restrict__ Qr,
                                                    _Float16* __restrict__ Kr,
                                                    _Float16* __restrict__ Vt) {
  int bt = blockIdx.x;          // 0 .. B*T-1
  int b = bt / Tt, t = bt % Tt;
  size_t src = (size_t)bt * Cc;
  for (int j = threadIdx.x; j < Cc / 2; j += blockDim.x) {
    float f  = freqs[(size_t)t * (Cc / 2) + j];
    float cs = __cosf(f), sn = __sinf(f);
    int c0 = 2 * j, c1 = c0 + 1;
    float q0 = Q[src + c0], q1 = Q[src + c1];
    float k0 = K[src + c0], k1 = K[src + c1];
    float qr0 = q0 * cs - q1 * sn, qr1 = q0 * sn + q1 * cs;
    float kr0 = k0 * cs - k1 * sn, kr1 = k0 * sn + k1 * cs;
    int h = c0 >> 6, d0 = c0 & 63, d1 = c1 & 63;
    size_t dst = ((size_t)(b * Hh + h) * Tt + t) * HD;
    Qr[dst + d0] = (_Float16)qr0; Qr[dst + d1] = (_Float16)qr1;
    Kr[dst + d0] = (_Float16)kr0; Kr[dst + d1] = (_Float16)kr1;
    size_t vrow = (size_t)(b * Hh + h) * HD;
    Vt[(vrow + d0) * Tt + t] = (_Float16)V[src + c0];
    Vt[(vrow + d1) * Tt + t] = (_Float16)V[src + c1];
  }
}

// ---------------------------------------------------------------------------
// Flash attention with pairwise bias + mask.
// Grid: (T/64, B*H); 128 threads = 4 waves; each wave owns 16 query rows.
// K/V tiles (32 keys) staged in LDS via global_load_async_to_lds_b128,
// double-buffered; waves share the staged tiles (removes 4x global redundancy).
// ---------------------------------------------------------------------------
#define KPAD 72   // K-tile row stride in halves (144B: 16B aligned, conflict-free)
#define VPAD 40   // V-tile row stride in halves (80B: 16B aligned, conflict-free)

__global__ __launch_bounds__(128) void attn_kernel(const _Float16* __restrict__ Qr,
                                                   const _Float16* __restrict__ Kr,
                                                   const _Float16* __restrict__ Vt,
                                                   const long long* __restrict__ qid,
                                                   const long long* __restrict__ kvid,
                                                   const unsigned char* __restrict__ amask,
                                                   const float* __restrict__ bias_emb,
                                                   _Float16* __restrict__ Y) {
  __shared__ __align__(16) _Float16 Kbuf[2][32][KPAD];  // 32 keys x 64 ch (padded)
  __shared__ __align__(16) _Float16 Vbuf[2][64][VPAD];  // 64 ch x 32 keys (padded)
  __shared__ __align__(16) _Float16 pbuf[4][16][40];    // per-wave 16x32 P tile

  int lane = threadIdx.x & 31;
  int wid  = threadIdx.x >> 5;
  int n  = lane & 15;    // C-layout column / B-fragment column
  int hb = lane >> 4;    // half-wave select
  int bh = blockIdx.y;
  int b = bh >> 4, h = bh & 15;
  int q0 = blockIdx.x * 64 + wid * 16;

  const _Float16* Qh = Qr + (size_t)bh * Tt * HD;
  const _Float16* Kh = Kr + (size_t)bh * Tt * HD;
  const _Float16* Vh = Vt + (size_t)bh * HD * Tt;
  _Float16* pw = &pbuf[wid][0][0];

  // Wave-relative LDS byte offsets for the async-copy destinations.
  unsigned kb[2] = { (unsigned)(uintptr_t)&Kbuf[0][0][0],
                     (unsigned)(uintptr_t)&Kbuf[1][0][0] };
  unsigned vb[2] = { (unsigned)(uintptr_t)&Vbuf[0][0][0],
                     (unsigned)(uintptr_t)&Vbuf[1][0][0] };

  // Issue one 32-key tile: K tile = 32 rows x 128B (8 chunks/row),
  // V tile = 64 rows x 64B (4 chunks/row); 256 16B chunks each; this wave
  // covers chunks [wid*64, wid*64+64) with 2 wave-instructions per tile.
  auto issue_tile = [&](int buf, int kt) {
    for (int t = 0; t < 2; t++) {
      int c = wid * 64 + t * 32 + lane;          // 0..255
      int kr = c >> 3, kw = c & 7;               // K: row, 16B chunk in row
      async_copy_16B(kb[buf] + (unsigned)(kr * KPAD * 2 + kw * 16),
                     Kh + (size_t)(kt + kr) * HD + kw * 8);
      int vr = c >> 2, vw = c & 3;               // V: row, 16B chunk in row
      async_copy_16B(vb[buf] + (unsigned)(vr * VPAD * 2 + vw * 16),
                     Vh + (size_t)vr * Tt + kt + vw * 8);
    }
  };

  // Q A-fragments for this wave's 16 rows (K-steps 0..31, 32..63) — invariant.
  v16h aq0 = load_a_frag(Qh + (size_t)q0 * HD + 0,  HD, lane);
  v16h aq1 = load_a_frag(Qh + (size_t)q0 * HD + 32, HD, lane);

  float bmiss  = bias_emb[h];
  float bmatch = bias_emb[Hh + h];
  long long qids[8];
  for (int v = 0; v < 8; v++) qids[v] = qid[(size_t)b * Tt + q0 + hb * 8 + v];

  float mrow[8], lrow[8];
  for (int v = 0; v < 8; v++) { mrow[v] = -1e30f; lrow[v] = 0.f; }
  v8f o[4] = {};   // output accumulators, 4 channel tiles of 16

  issue_tile(0, 0);   // prologue

  int cur = 0;
  for (int kt = 0; kt < Tt; kt += 32, cur ^= 1) {
    // Own async loads for buffer `cur` landed; barrier => everyone's landed,
    // and everyone finished reading the other buffer (s_wait_dscnt below).
    asm volatile("s_wait_asynccnt 0x0" ::: "memory");
    __syncthreads();
    if (kt + 32 < Tt) issue_tile(cur ^ 1, kt + 32);

    // Scores: S[16q x 32k], K operand from LDS tile.
    v16h bk00 = *(const v16h*)&Kbuf[cur][n][hb * 16];
    v16h bk01 = *(const v16h*)&Kbuf[cur][n][32 + hb * 16];
    v16h bk10 = *(const v16h*)&Kbuf[cur][16 + n][hb * 16];
    v16h bk11 = *(const v16h*)&Kbuf[cur][16 + n][32 + hb * 16];
    v8f s0 = {}, s1 = {};
    s0 = WMMA(aq0, bk00, s0); s0 = WMMA(aq1, bk01, s0);
    s1 = WMMA(aq0, bk10, s1); s1 = WMMA(aq1, bk11, s1);

    long long kid0 = kvid[(size_t)b * Tt + kt + n];
    long long kid1 = kvid[(size_t)b * Tt + kt + 16 + n];
    bool mk0 = amask[(size_t)b * Tt + kt + n] != 0;
    bool mk1 = amask[(size_t)b * Tt + kt + 16 + n] != 0;

    float p0[8], p1[8], rmax[8];
    for (int v = 0; v < 8; v++) {
      float x0 = s0[v] * 0.125f + ((qids[v] == kid0) ? bmatch : bmiss);
      float x1 = s1[v] * 0.125f + ((qids[v] == kid1) ? bmatch : bmiss);
      if (!mk0) x0 = -1e30f;
      if (!mk1) x1 = -1e30f;
      p0[v] = x0; p1[v] = x1;
      float r = fmaxf(x0, x1);                 // row-max across 16-lane half
      r = fmaxf(r, __shfl_xor(r, 1, 32));
      r = fmaxf(r, __shfl_xor(r, 2, 32));
      r = fmaxf(r, __shfl_xor(r, 4, 32));
      r = fmaxf(r, __shfl_xor(r, 8, 32));
      rmax[v] = r;
    }
    for (int v = 0; v < 8; v++) {
      float mnew  = fmaxf(mrow[v], rmax[v]);
      float alpha = __expf(mrow[v] - mnew);
      float e0 = __expf(p0[v] - mnew);
      float e1 = __expf(p1[v] - mnew);
      p0[v] = e0; p1[v] = e1;
      float rs = e0 + e1;
      rs += __shfl_xor(rs, 1, 32);
      rs += __shfl_xor(rs, 2, 32);
      rs += __shfl_xor(rs, 4, 32);
      rs += __shfl_xor(rs, 8, 32);
      lrow[v] = lrow[v] * alpha + rs;
      mrow[v] = mnew;
      for (int ct = 0; ct < 4; ct++) o[ct][v] *= alpha;
    }

    // Re-lay P from C-layout into A-fragment layout via this wave's LDS tile.
    for (int v = 0; v < 8; v++) {
      pw[(v + 8 * hb) * 40 + n]      = (_Float16)p0[v];
      pw[(v + 8 * hb) * 40 + 16 + n] = (_Float16)p1[v];
    }
    asm volatile("s_wait_dscnt 0x0" ::: "memory");   // intra-wave cross-lane RAW
    union { v16h v; v8h x[2]; } up;
    up.x[0] = *(const v8h*)(pw + n * 40 + hb * 16);
    up.x[1] = *(const v8h*)(pw + n * 40 + hb * 16 + 8);
    v16h pa = up.v;

    // O += P @ V ; V operand from LDS tile (channels-major).
    for (int ct = 0; ct < 4; ct++) {
      v16h bv = *(const v16h*)&Vbuf[cur][ct * 16 + n][hb * 16];
      o[ct] = WMMA(pa, bv, o[ct]);
    }

    // All of this wave's LDS reads complete before next barrier (WAR guard
    // against async refill of the buffer we just consumed).
    asm volatile("s_wait_dscnt 0x0" ::: "memory");
  }

  // Normalize and store y (B,T,C) f16.
  for (int v = 0; v < 8; v++) {
    float inv = (lrow[v] > 0.f) ? 1.0f / lrow[v] : 0.f;
    int row = q0 + 8 * hb + v;
    size_t dst = ((size_t)b * Tt + row) * Cc + h * HD;
    for (int ct = 0; ct < 4; ct++)
      Y[dst + ct * 16 + n] = (_Float16)(o[ct][v] * inv);
  }
}

// ---------------------------------------------------------------------------
// Host-side orchestration.
// ---------------------------------------------------------------------------
extern "C" void kernel_launch(void* const* d_in, const int* in_sizes, int n_in,
                              void* d_out, int out_size, void* d_ws, size_t ws_size,
                              hipStream_t stream) {
  const float*      q     = (const float*)d_in[0];
  const float*      k     = (const float*)d_in[1];
  const float*      v     = (const float*)d_in[2];
  const float*      freqs = (const float*)d_in[3];
  const long long*  qid   = (const long long*)d_in[4];
  const long long*  kvid  = (const long long*)d_in[5];
  const unsigned char* am = (const unsigned char*)d_in[6];
  const float*      Wq    = (const float*)d_in[7];
  const float*      Wk    = (const float*)d_in[8];
  const float*      Wv    = (const float*)d_in[9];
  const float*      Wo    = (const float*)d_in[10];
  const float*      bias  = (const float*)d_in[11];
  float* out = (float*)d_out;

  const size_t NA = (size_t)BT * Cc;   // 4M activation elements
  const size_t NW = (size_t)Cc * Cc;   // 1M weight elements

  _Float16* qf  = (_Float16*)d_ws;
  _Float16* kf  = qf  + NA;
  _Float16* vf  = kf  + NA;
  _Float16* wqf = vf  + NA;
  _Float16* wkf = wqf + NW;
  _Float16* wvf = wkf + NW;
  _Float16* wof = wvf + NW;
  float* Qf32 = (float*)(wof + NW);
  float* Kf32 = Qf32 + NA;
  float* Vf32 = Kf32 + NA;
  _Float16* Qr = (_Float16*)(Vf32 + NA);
  _Float16* Kr = Qr + NA;
  _Float16* Vt = Kr + NA;
  _Float16* yf = (_Float16*)Qf32;      // Qf32 is dead after rope_reshape

  // fp32 -> f16 conversions
  cvt_f32_f16<<<(int)(NA / 4 / 256), 256, 0, stream>>>((const float4*)q,  (v4h*)qf,  (int)(NA / 4));
  cvt_f32_f16<<<(int)(NA / 4 / 256), 256, 0, stream>>>((const float4*)k,  (v4h*)kf,  (int)(NA / 4));
  cvt_f32_f16<<<(int)(NA / 4 / 256), 256, 0, stream>>>((const float4*)v,  (v4h*)vf,  (int)(NA / 4));
  cvt_f32_f16<<<(int)(NW / 4 / 256), 256, 0, stream>>>((const float4*)Wq, (v4h*)wqf, (int)(NW / 4));
  cvt_f32_f16<<<(int)(NW / 4 / 256), 256, 0, stream>>>((const float4*)Wk, (v4h*)wkf, (int)(NW / 4));
  cvt_f32_f16<<<(int)(NW / 4 / 256), 256, 0, stream>>>((const float4*)Wv, (v4h*)wvf, (int)(NW / 4));
  cvt_f32_f16<<<(int)(NW / 4 / 256), 256, 0, stream>>>((const float4*)Wo, (v4h*)wof, (int)(NW / 4));

  // Projections (WMMA)
  dim3 gg(Cc / 128, BT / 64);
  gemm_xwT<<<gg, 256, 0, stream>>>(qf, wqf, Qf32, BT, Cc, Cc);
  gemm_xwT<<<gg, 256, 0, stream>>>(kf, wkf, Kf32, BT, Cc, Cc);
  gemm_xwT<<<gg, 256, 0, stream>>>(vf, wvf, Vf32, BT, Cc, Cc);

  // RoPE + head reshape (+ V transpose)
  rope_reshape<<<BT, 256, 0, stream>>>(Qf32, Kf32, Vf32, freqs, Qr, Kr, Vt);

  // Attention (WMMA flash, async-LDS staged K/V)
  dim3 ga(Tt / 64, Bb * Hh);
  attn_kernel<<<ga, 128, 0, stream>>>(Qr, Kr, Vt, qid, kvid, am, bias, yf);

  // Output projection (WMMA), f32 straight to d_out
  gemm_xwT<<<gg, 256, 0, stream>>>(yf, wof, out, BT, Cc, Cc);
}
